// GRU_28561532518806
// MI455X (gfx1250) — matrix operand order repaired and checked by reading, not compiled, not run on hardware
//
#include <hip/hip_runtime.h>
#include <hip/hip_bf16.h>

typedef __attribute__((ext_vector_type(16))) _Float16     v16h;
typedef __attribute__((ext_vector_type(8)))  _Float16     v8h;
typedef __attribute__((ext_vector_type(8)))  float        v8f;
typedef __attribute__((ext_vector_type(4)))  unsigned int v4u;
typedef __attribute__((ext_vector_type(8)))  int          v8i;
typedef __attribute__((ext_vector_type(4)))  int          v4i;

static constexpr int SEQL  = 512;
static constexpr int BATCH = 64;
static constexpr int IDIM  = 512;
static constexpr int HDIM  = 1024;
static constexpr int ODIM  = 512;
static constexpr int G3    = 3 * HDIM;   // z,r,h gates concatenated

// Tensor Data Mover path (gfx1250): guarded so host pass / older toolchains fall
// back to the synchronous LDS staging path.
#if __has_builtin(__builtin_amdgcn_tensor_load_to_lds) && \
    __has_builtin(__builtin_amdgcn_s_wait_tensorcnt)
#define GRU_USE_TDM 1
#else
#define GRU_USE_TDM 0
#endif

// ---------------------------------------------------------------------------
// fp32 -> f16 conversion / packing
// ---------------------------------------------------------------------------
__global__ void f32_to_f16_kernel(const float* __restrict__ src,
                                  _Float16* __restrict__ dst, int n) {
    int i = blockIdx.x * blockDim.x + threadIdx.x;
    if (i < n) dst[i] = (_Float16)src[i];
}

__global__ void init_h_kernel(_Float16* __restrict__ h16,
                              float* __restrict__ hf, int n) {
    int i = blockIdx.x * blockDim.x + threadIdx.x;
    if (i < n) { h16[i] = (_Float16)0.0f; hf[i] = 0.0f; }
}

#if GRU_USE_TDM
// ---------------------------------------------------------------------------
// Issue one TDM descriptor moving a [128 rows x 32 halves] tile from global
// (row stride = ld_elems halves) into LDS with an 8-half pad per row, i.e. the
// [128][40] padded layout the compute waves read. data_size=2B.
//   group0: count=1 | lds_addr | global_addr(57b) | type=2
//   group1: data_size=1, pad_enable, pad_interval=3 (16 DW), pad_amount=3 (4 DW),
//           tensor_dim0=ld, tensor_dim1=rows_total, tile_dim0=32, tile_dim1=128,
//           tensor_dim0_stride=ld
// ---------------------------------------------------------------------------
__device__ inline void tdm_load_tile_128x32(const _Float16* gptr, int ld_elems,
                                            unsigned int lds_addr, int rows_total) {
    unsigned long long ga = (unsigned long long)(uintptr_t)gptr;
    v4u g0 = { 1u,                                   // count = 1 valid descriptor
               lds_addr,                             // LDS byte address
               (unsigned int)ga,                     // global_addr[31:0]
               (unsigned int)((ga >> 32) & 0x01FFFFFFu) | 0x80000000u }; // [56:32] | type=2
    unsigned int ldu = (unsigned int)ld_elems;
    unsigned int rtu = (unsigned int)rows_total;
    v8i g1 = { (int)((1u << 16) |                    // data_size = 2 bytes
                     (1u << 20) |                    // pad_enable
                     (3u << 22) |                    // pad_interval: 16 DWORDs (64B row)
                     (3u << 25)),                    // pad_amount:   4 DWORDs (8 halves)
               (int)((ldu & 0xFFFFu) << 16),         // tensor_dim0 low16 (atomic_barrier=0)
               (int)((ldu >> 16) | ((rtu & 0xFFFFu) << 16)),   // dim0 hi | dim1 lo
               (int)((rtu >> 16) | (32u << 16)),     // dim1 hi | tile_dim0 = 32
               (int)128u,                            // tile_dim1 = 128, tile_dim2 = 0
               (int)ldu,                             // tensor_dim0_stride low32
               0, 0 };                               // stride hi / dim1_stride = 0
    v4i gz = { 0, 0, 0, 0 };
#if defined(__clang_major__) && (__clang_major__ >= 23)
    v8i gz8 = {};
    __builtin_amdgcn_tensor_load_to_lds(g0, g1, gz, gz, gz8, 0);
#else
    __builtin_amdgcn_tensor_load_to_lds(g0, g1, gz, gz, 0);
#endif
}
#endif // GRU_USE_TDM

// Compute 8 WMMAs (2 m-tiles x 4 n-tiles) from one staged 128x32 tile pair.
#define GEMM_COMPUTE_TILE(SA, SB)                                              \
    do {                                                                       \
        v16h afrag[2];                                                         \
        _Pragma("unroll")                                                      \
        for (int mt = 0; mt < 2; ++mt) {                                       \
            int row = wm * 32 + mt * 16 + r;                                   \
            v8h lo = *(const v8h*)&SA[row][hiL * 8];                           \
            v8h hh = *(const v8h*)&SA[row][16 + hiL * 8];                      \
            _Pragma("unroll")                                                  \
            for (int i = 0; i < 8; ++i) {                                      \
                afrag[mt][i] = lo[i]; afrag[mt][i + 8] = hh[i];                \
            }                                                                  \
        }                                                                      \
        v16h bfrag[4];                                                         \
        _Pragma("unroll")                                                      \
        for (int nt = 0; nt < 4; ++nt) {                                       \
            int row = wn * 64 + nt * 16 + r;                                   \
            v8h lo = *(const v8h*)&SB[row][hiL * 8];                           \
            v8h hh = *(const v8h*)&SB[row][16 + hiL * 8];                      \
            _Pragma("unroll")                                                  \
            for (int i = 0; i < 8; ++i) {                                      \
                bfrag[nt][i] = lo[i]; bfrag[nt][i + 8] = hh[i];                \
            }                                                                  \
        }                                                                      \
        _Pragma("unroll")                                                      \
        for (int mt = 0; mt < 2; ++mt)                                         \
            _Pragma("unroll")                                                  \
            for (int nt = 0; nt < 4; ++nt)                                     \
                acc[mt][nt] = __builtin_amdgcn_wmma_f32_16x16x32_f16(          \
                    false, afrag[mt], false, bfrag[nt],                        \
                    (short)0, acc[mt][nt], false, false);                      \
    } while (0)

// ---------------------------------------------------------------------------
// Tiled WMMA GEMM:  C[M,N] = A[M,K] (f16, row-major) * W[N,K]^T (f16, row-major)
// Block tile 128x128, K-step 32, 256 threads = 8 waves (4x2), wave tile 32x64.
// A/B tiles staged to LDS by the Tensor Data Mover (double-buffered,
// TENSORcnt-synchronized); falls back to cooperative vector loads.
// OUT_F32: write fp32 + bias (final projection); else write f16 (intermediates).
// ---------------------------------------------------------------------------
template<bool OUT_F32>
__global__ __launch_bounds__(256)
void gemm_xw_t(const _Float16* __restrict__ A, int lda, int mTot,
               const _Float16* __restrict__ W, int ldw, int nTot,
               const float* __restrict__ bias,
               void* __restrict__ Cout, int ldc, int K) {
#if GRU_USE_TDM
    __shared__ _Float16 sA[2][128][40];   // +8 halves pad (matches TDM pad_amount)
    __shared__ _Float16 sB[2][128][40];
#else
    __shared__ _Float16 sA[1][128][40];
    __shared__ _Float16 sB[1][128][40];
#endif

    const int tid  = threadIdx.x;
    const int lane = tid & 31;
    const int wave = tid >> 5;
    const int wm   = wave >> 1;          // 0..3
    const int wn   = wave & 1;           // 0..1
    const int r    = lane & 15;
    const int hiL  = lane >> 4;          // lane half (wave32)
    const int rowBase = blockIdx.y * 128;
    const int colBase = blockIdx.x * 128;

    v8f acc[2][4];
    #pragma unroll
    for (int mt = 0; mt < 2; ++mt)
        #pragma unroll
        for (int nt = 0; nt < 4; ++nt) { v8f z = {}; acc[mt][nt] = z; }

#if GRU_USE_TDM
    const int nk = K / 32;
    const unsigned int aAddr[2] = { (unsigned int)(uintptr_t)&sA[0][0][0],
                                    (unsigned int)(uintptr_t)&sA[1][0][0] };
    const unsigned int bAddr[2] = { (unsigned int)(uintptr_t)&sB[0][0][0],
                                    (unsigned int)(uintptr_t)&sB[1][0][0] };
    if (tid == 0) {
        tdm_load_tile_128x32(A + (size_t)rowBase * lda, lda, aAddr[0], mTot);
        tdm_load_tile_128x32(W + (size_t)colBase * ldw, ldw, bAddr[0], nTot);
        __builtin_amdgcn_s_wait_tensorcnt(0);
    }
    __syncthreads();
    for (int kt = 0; kt < nk; ++kt) {
        const int cur = kt & 1;
        const int nxt = cur ^ 1;
        if (tid == 0 && (kt + 1) < nk) {   // DMA next tile while waves compute
            int k1 = (kt + 1) * 32;
            tdm_load_tile_128x32(A + (size_t)rowBase * lda + k1, lda, aAddr[nxt], mTot);
            tdm_load_tile_128x32(W + (size_t)colBase * ldw + k1, ldw, bAddr[nxt], nTot);
        }
        GEMM_COMPUTE_TILE(sA[cur], sB[cur]);
        if (tid == 0) __builtin_amdgcn_s_wait_tensorcnt(0);
        __syncthreads();
    }
#else
    const int tRow = tid >> 1;           // 0..127
    const int tCol = (tid & 1) * 16;     // 0 or 16
    for (int k0 = 0; k0 < K; k0 += 32) {
        {
            const _Float16* ga = A + (size_t)(rowBase + tRow) * lda + k0 + tCol;
            *(v8h*)&sA[0][tRow][tCol]     = *(const v8h*)ga;
            *(v8h*)&sA[0][tRow][tCol + 8] = *(const v8h*)(ga + 8);
            const _Float16* gb = W + (size_t)(colBase + tRow) * ldw + k0 + tCol;
            *(v8h*)&sB[0][tRow][tCol]     = *(const v8h*)gb;
            *(v8h*)&sB[0][tRow][tCol + 8] = *(const v8h*)(gb + 8);
        }
        __syncthreads();
        GEMM_COMPUTE_TILE(sA[0], sB[0]);
        __syncthreads();
    }
#endif

    // Epilogue. C/D layout: VGPR v -> M = v + 8*laneHalf, N = lane&15.
    float*    Cf = (float*)Cout;
    _Float16* Ch = (_Float16*)Cout;
    #pragma unroll
    for (int mt = 0; mt < 2; ++mt) {
        #pragma unroll
        for (int nt = 0; nt < 4; ++nt) {
            int gr0 = rowBase + wm * 32 + mt * 16 + hiL * 8;
            int gc  = colBase + wn * 64 + nt * 16 + r;
            v8f c = acc[mt][nt];
            #pragma unroll
            for (int v = 0; v < 8; ++v) {
                size_t off = (size_t)(gr0 + v) * ldc + gc;
                if (OUT_F32) Cf[off] = c[v] + (bias ? bias[gc] : 0.0f);
                else         Ch[off] = (_Float16)c[v];
            }
        }
    }
}

// ---------------------------------------------------------------------------
// One GRU timestep.  grid = H/16 blocks, 384 threads = 12 waves.
// wave = (gate in 0..2) x (batch m-tile in 0..3); each wave: one 16x16
// accumulator over K = HDIM via v_wmma_f32_16x16x32_f16 (weights L2-resident,
// 6 MB << 192 MB L2, re-read 512x).  Then fused gate nonlinearities + h update.
// h is ping-pong double-buffered across launches (implicit grid sync).
// ---------------------------------------------------------------------------
__global__ __launch_bounds__(384)
void gru_step_kernel(const _Float16* __restrict__ h16,  // prev h, f16 [B,H]
                     const float* __restrict__ hf,      // prev h, f32 [B,H]
                     _Float16* __restrict__ h16n,       // next h, f16
                     float* __restrict__ hfn,           // next h, f32
                     const _Float16* __restrict__ Whcat,// [3H, H] row-major
                     const _Float16* __restrict__ xg_s, // [B, 3H] this step
                     const float* __restrict__ bxz, const float* __restrict__ bhz,
                     const float* __restrict__ bxr, const float* __restrict__ bhr,
                     const float* __restrict__ bxh, const float* __restrict__ bhh,
                     _Float16* __restrict__ hs_s) {     // [B, H] this step
    __shared__ float sg[3][64][16];

    const int tid  = threadIdx.x;
    const int lane = tid & 31;
    const int wave = tid >> 5;       // 0..11
    const int gate = wave >> 2;      // 0..2 (z,r,h)
    const int mt   = wave & 3;       // batch tile 0..3
    const int r    = lane & 15;
    const int hiL  = lane >> 4;
    const int n0   = blockIdx.x * 16;

    v8f acc = {};
    const _Float16* arow = h16   + (size_t)(mt * 16 + r) * HDIM;
    const _Float16* wrow = Whcat + (size_t)(gate * HDIM + n0 + r) * HDIM;

    #pragma unroll 4
    for (int k0 = 0; k0 < HDIM; k0 += 32) {
        v8h alo = *(const v8h*)(arow + k0 + hiL * 8);
        v8h ahi = *(const v8h*)(arow + k0 + 16 + hiL * 8);
        v8h blo = *(const v8h*)(wrow + k0 + hiL * 8);
        v8h bhi = *(const v8h*)(wrow + k0 + 16 + hiL * 8);
        v16h a, b;
        #pragma unroll
        for (int i = 0; i < 8; ++i) {
            a[i] = alo[i]; a[i + 8] = ahi[i];
            b[i] = blo[i]; b[i + 8] = bhi[i];
        }
        acc = __builtin_amdgcn_wmma_f32_16x16x32_f16(
            false, a, false, b, (short)0, acc, false, false);
    }

    #pragma unroll
    for (int v = 0; v < 8; ++v)
        sg[gate][mt * 16 + hiL * 8 + v][r] = acc[v];
    __syncthreads();

    // fused elementwise GRU update for the 64x16 slab
    for (int idx = tid; idx < BATCH * 16; idx += 384) {
        int b  = idx >> 4;
        int n  = idx & 15;
        int gn = n0 + n;
        float xz = (float)xg_s[b * G3 + 0 * HDIM + gn] + bxz[gn];
        float xr = (float)xg_s[b * G3 + 1 * HDIM + gn] + bxr[gn];
        float xh = (float)xg_s[b * G3 + 2 * HDIM + gn] + bxh[gn];
        float dz = sg[0][b][n] + bhz[gn];
        float dr = sg[1][b][n] + bhr[gn];
        float dh = sg[2][b][n] + bhh[gn];
        float z  = 1.0f / (1.0f + __expf(-(xz + dz)));
        float rr = 1.0f / (1.0f + __expf(-(xr + dr)));
        float hc = tanhf(xh + rr * dh);
        float hp = hf[b * HDIM + gn];
        float hn = (1.0f - z) * hp + z * hc;
        hfn [b * HDIM + gn] = hn;
        h16n[b * HDIM + gn] = (_Float16)hn;
        hs_s[b * HDIM + gn] = (_Float16)hn;
    }
}

// ---------------------------------------------------------------------------
extern "C" void kernel_launch(void* const* d_in, const int* in_sizes, int n_in,
                              void* d_out, int out_size, void* d_ws, size_t ws_size,
                              hipStream_t stream) {
    (void)in_sizes; (void)n_in; (void)out_size; (void)ws_size;

    const float* x   = (const float*)d_in[0];
    const float* Wxz = (const float*)d_in[1];
    const float* bxz = (const float*)d_in[2];
    const float* Whz = (const float*)d_in[3];
    const float* bhz = (const float*)d_in[4];
    const float* Wxr = (const float*)d_in[5];
    const float* bxr = (const float*)d_in[6];
    const float* Whr = (const float*)d_in[7];
    const float* bhr = (const float*)d_in[8];
    const float* Wxh = (const float*)d_in[9];
    const float* bxh = (const float*)d_in[10];
    const float* Whh = (const float*)d_in[11];
    const float* bhh = (const float*)d_in[12];
    const float* Why = (const float*)d_in[13];
    const float* bhy = (const float*)d_in[14];

    // workspace carve (256B aligned), total ~313 MB
    char* p = (char*)d_ws;
    auto carve = [&](size_t bytes) -> void* {
        void* q = (void*)p;
        p += (bytes + 255) & ~(size_t)255;
        return q;
    };
    _Float16* xh   = (_Float16*)carve((size_t)SEQL * BATCH * IDIM * 2);
    _Float16* Wxc  = (_Float16*)carve((size_t)G3 * IDIM * 2);
    _Float16* Whc  = (_Float16*)carve((size_t)G3 * HDIM * 2);
    _Float16* Whyh = (_Float16*)carve((size_t)ODIM * HDIM * 2);
    _Float16* xg   = (_Float16*)carve((size_t)SEQL * BATCH * G3 * 2);
    _Float16* hs   = (_Float16*)carve((size_t)SEQL * BATCH * HDIM * 2);
    _Float16* h16b[2];
    h16b[0] = (_Float16*)carve((size_t)BATCH * HDIM * 2);
    h16b[1] = (_Float16*)carve((size_t)BATCH * HDIM * 2);
    float* hfb[2];
    hfb[0] = (float*)carve((size_t)BATCH * HDIM * 4);
    hfb[1] = (float*)carve((size_t)BATCH * HDIM * 4);

    auto cvt = [&](const float* s, _Float16* d, size_t n) {
        f32_to_f16_kernel<<<dim3((unsigned)((n + 255) / 256)), dim3(256), 0, stream>>>(
            s, d, (int)n);
    };

    // pack inputs/weights to f16 (gate order z, r, h)
    cvt(x,   xh,                                 (size_t)SEQL * BATCH * IDIM);
    cvt(Wxz, Wxc + (size_t)0 * HDIM * IDIM,      (size_t)HDIM * IDIM);
    cvt(Wxr, Wxc + (size_t)1 * HDIM * IDIM,      (size_t)HDIM * IDIM);
    cvt(Wxh, Wxc + (size_t)2 * HDIM * IDIM,      (size_t)HDIM * IDIM);
    cvt(Whz, Whc + (size_t)0 * HDIM * HDIM,      (size_t)HDIM * HDIM);
    cvt(Whr, Whc + (size_t)1 * HDIM * HDIM,      (size_t)HDIM * HDIM);
    cvt(Whh, Whc + (size_t)2 * HDIM * HDIM,      (size_t)HDIM * HDIM);
    cvt(Why, Whyh,                               (size_t)ODIM * HDIM);

    // phase 1: xg[s,b,:] = x[s,b,:] @ [Wxz;Wxr;Wxh]^T   (biases folded in later)
    gemm_xw_t<false><<<dim3(G3 / 128, (SEQL * BATCH) / 128), 256, 0, stream>>>(
        xh, IDIM, SEQL * BATCH, Wxc, IDIM, G3, nullptr, (void*)xg, G3, IDIM);

    // h0 = 0
    init_h_kernel<<<dim3((BATCH * HDIM) / 256), 256, 0, stream>>>(
        h16b[0], hfb[0], BATCH * HDIM);

    // phase 2: sequential scan, one launch per step (graph-captured)
    for (int s = 0; s < SEQL; ++s) {
        int cur = s & 1, nxt = cur ^ 1;
        gru_step_kernel<<<dim3(HDIM / 16), 384, 0, stream>>>(
            h16b[cur], hfb[cur], h16b[nxt], hfb[nxt],
            Whc, xg + (size_t)s * BATCH * G3,
            bxz, bhz, bxr, bhr, bxh, bhh,
            hs + (size_t)s * BATCH * HDIM);
    }

    // phase 3: out[s,b,:] = hs[s,b,:] @ Why^T + bhy  (fp32 output)
    gemm_xw_t<true><<<dim3(ODIM / 128, (SEQL * BATCH) / 128), 256, 0, stream>>>(
        hs, HDIM, SEQL * BATCH, Whyh, HDIM, ODIM, bhy, d_out, ODIM, HDIM);
}
